// HRNN_4844723110361
// MI455X (gfx1250) — compile-verified, matrix operand order checked by reading, not run on hardware
//
#include <hip/hip_runtime.h>
#include <hip/hip_bf16.h>

// ---------------------------------------------------------------------------
// MI455X (gfx1250) implementation.
//
// Roofline: ~185M weight elements touched on every one of 16 sequential
// timesteps with M = batch = 4 (AI ~ 4 MAC/weight) => bound by weight
// streaming at 23.3 TB/s. Therefore:
//   * Convert all weights fp32 -> bf16 once per launch, pre-swizzled into
//     WMMA B-fragment order (halves HBM traffic; one coalesced 32B load per
//     lane per fragment), compute with v_wmma_f32_16x16x32_bf16 (f32 accum).
//   * GEMM kernel: wave32, each wave owns TWO 16-wide N tiles sharing one
//     A fragment (amortizes fp32->bf16 A packing, 2 WMMAs per A build).
//     Batch rows 0..3 replicated into WMMA rows 4..15 (outputs discarded).
//     Two-segment A handles all concat[] inputs without materializing.
//   * Decoder layers (M=4, N<=2048 => too few tiles to hide HBM latency) use
//     grid.z split-K (S=8): z-slices write fp32 partials, a reduce kernel
//     sums + bias + activation + dual-destination store.
//   * Recurrence via stream-ordered launches (graph-captured by harness).
// ---------------------------------------------------------------------------

typedef __attribute__((ext_vector_type(16))) __bf16         v16bf;
typedef __attribute__((ext_vector_type(16))) unsigned short v16u;
typedef __attribute__((ext_vector_type(8)))  float          v8f;
typedef __attribute__((ext_vector_type(4)))  float          v4f;

__device__ __forceinline__ unsigned short f2bf(float f) {
  unsigned int u = __builtin_bit_cast(unsigned int, f);
  u += 0x7FFFu + ((u >> 16) & 1u);              // round to nearest even
  return (unsigned short)(u >> 16);
}

__device__ __forceinline__ float gelu_f(float x) {
  float x3 = x * x * x;
  return 0.5f * x * (1.0f + tanhf(0.7978845608028654f * (x + 0.044715f * x3)));
}

__global__ __launch_bounds__(256) void zero_f32_kernel(float* p, int n) {
  int i = blockIdx.x * blockDim.x + threadIdx.x;
  if (i < n) p[i] = 0.0f;
}

// Reorder a (K,N) fp32 row-major weight matrix into bf16 WMMA B-fragment
// order. For tile (kt,nt), lane l (n = nt*16 + (l&15), hi = l>>4):
//   elems 0..7  <- W[kt*32 + hi*8      + j][n]
//   elems 8..15 <- W[kt*32 + 16 + hi*8 + j][n]
// Fragments stored contiguously: dst[(((kt*NT)+nt)*32 + l)*16 + j].
__global__ __launch_bounds__(256) void swz_kernel(const float* __restrict__ W,
                                                  unsigned short* __restrict__ dst,
                                                  int K, int N, long long wE) {
  int e = blockIdx.y;
  int NT = N >> 4;
  long long idx = (long long)blockIdx.x * blockDim.x + threadIdx.x;
  long long total = (long long)(K >> 5) * NT * 32;
  if (idx >= total) return;
  int l  = (int)(idx & 31);
  long long tt = idx >> 5;
  int nt = (int)(tt % NT);
  int kt = (int)(tt / NT);
  int n  = nt * 16 + (l & 15);
  int hi = l >> 4;
  const float* w = W + (long long)e * wE;
  unsigned short* d = dst + (long long)e * wE + idx * 16;
  int k0 = kt * 32 + hi * 8;
#pragma unroll
  for (int j = 0; j < 8; ++j) d[j]     = f2bf(w[(long long)(k0 + j) * N + n]);
#pragma unroll
  for (int j = 0; j < 8; ++j) d[8 + j] = f2bf(w[(long long)(k0 + 16 + j) * N + n]);
}

// Sum split-K partials, add bias, activate, dual-destination store.
// part layout: [S][4][N] fp32.
__global__ __launch_bounds__(256) void reduce_kernel(
    const float* __restrict__ part, int S, int N,
    const float* __restrict__ bias,
    float* __restrict__ out, int oLd,
    float* __restrict__ out2, int o2Ld, int act)
{
  int i = blockIdx.x * blockDim.x + threadIdx.x;
  if (i >= 4 * N) return;
  int m = i / N, n = i - m * N;
  float v = bias ? bias[n] : 0.0f;
  for (int s = 0; s < S; ++s) v += part[(long long)s * 4 * N + m * N + n];
  if (act == 1)      v = gelu_f(v);
  else if (act == 2) v = tanhf(v);
  out[(long long)m * oLd + n] = v;
  if (out2) out2[(long long)m * o2Ld + n] = v;
}

// Generic small-M GEMM: out[e, m, :] = act( A[e, m, :] @ W[e] + bias[e] )
// A = concat of two fp32 row-major segments (K0, K1 cols). W pre-swizzled
// bf16. Each wave: 2 N-tiles, one shared A fragment per K-chunk.
// splitK > 0: blockIdx.z selects a K-chunk of splitK columns; raw fp32
// partials are written to out[ks][4][N] (bias/act deferred to reduce).
__global__ __launch_bounds__(256) void gemm_bf16_kernel(
    const float* __restrict__ A0, long long a0e, int a0ld, int K0,
    const float* __restrict__ A1, long long a1e, int a1ld, int K1,
    const unsigned short* __restrict__ W, long long wE,
    const float* __restrict__ bias, long long bE,
    float* __restrict__ out,  long long oE,  int oLd,
    float* __restrict__ out2, long long o2E, int o2Ld,
    int N, int act, int splitK)
{
  const int e    = blockIdx.y;
  const int wave = threadIdx.x >> 5;
  const int lane = threadIdx.x & 31;
  const int NT   = N >> 4;
  const int nt0  = (blockIdx.x * 8 + wave) * 2;
  if (nt0 >= NT) return;
  const int ks     = (splitK > 0) ? blockIdx.z : 0;
  const int kShift = ks * splitK;
  const int KT     = ((splitK > 0) ? splitK : (K0 + K1)) >> 5;
  const int n15    = lane & 15;
  const int hi     = lane >> 4;
  const int mrow   = n15 & 3;  // replicate batch rows 0..3 into rows 4..15

  const float* a0 = A0 + (long long)e * a0e + (long long)mrow * a0ld;
  const float* a1 = (K1 > 0) ? (A1 + (long long)e * a1e + (long long)mrow * a1ld)
                             : nullptr;
  const unsigned short* wp = W + (long long)e * wE +
      (((long long)(kShift >> 5) * NT + nt0) * 32 + lane) * 16;
  const long long wStep = (long long)NT * 512;  // 32 lanes * 16 elems / ktile

  v8f acc0 = {}, acc1 = {};
  for (int kt = 0; kt < KT; ++kt) {
    const int kb = kShift + kt * 32 + hi * 8;
    v16u au;
    {   // run 0: K = kb .. kb+7 (contiguous fp32, never straddles K0)
      const float* p = (kb < K0) ? (a0 + kb) : (a1 + (kb - K0));
      v4f x0 = *(const v4f*)p;
      v4f x1 = *(const v4f*)(p + 4);
#pragma unroll
      for (int j = 0; j < 4; ++j) au[j]     = f2bf(x0[j]);
#pragma unroll
      for (int j = 0; j < 4; ++j) au[4 + j] = f2bf(x1[j]);
    }
    {   // run 1: K = kb+16 .. kb+23
      const int k = kb + 16;
      const float* p = (k < K0) ? (a0 + k) : (a1 + (k - K0));
      v4f x0 = *(const v4f*)p;
      v4f x1 = *(const v4f*)(p + 4);
#pragma unroll
      for (int j = 0; j < 4; ++j) au[8 + j]  = f2bf(x0[j]);
#pragma unroll
      for (int j = 0; j < 4; ++j) au[12 + j] = f2bf(x1[j]);
    }
    v16u bu0 = *(const v16u*)wp;            // tile nt0   (32B coalesced)
    v16u bu1 = *(const v16u*)(wp + 512);    // tile nt0+1
    if (kt + 1 < KT) __builtin_prefetch(wp + wStep, 0, 1);
    v16bf a  = __builtin_bit_cast(v16bf, au);
    acc0 = __builtin_amdgcn_wmma_f32_16x16x32_bf16(
        false, a, false, __builtin_bit_cast(v16bf, bu0), (short)0, acc0, false, false);
    acc1 = __builtin_amdgcn_wmma_f32_16x16x32_bf16(
        false, a, false, __builtin_bit_cast(v16bf, bu1), (short)0, acc1, false, false);
    wp += wStep;
  }

  // D layout: lanes 0..15 (hi==0) hold rows M=0..7 in acc[0..7]; rows 0..3
  // are the real batch rows.
  if (hi == 0) {
#pragma unroll
    for (int j = 0; j < 2; ++j) {
      const v8f& acc = j ? acc1 : acc0;
      const int col = (nt0 + j) * 16 + n15;
      if (splitK > 0) {
#pragma unroll
        for (int r = 0; r < 4; ++r)
          out[(long long)ks * 4 * N + (long long)r * N + col] = acc[r];
      } else {
        const float bv = bias ? bias[(long long)e * bE + col] : 0.0f;
#pragma unroll
        for (int r = 0; r < 4; ++r) {
          float v = acc[r] + bv;
          if (act == 1)      v = gelu_f(v);
          else if (act == 2) v = tanhf(v);
          out[(long long)e * oE + (long long)r * oLd + col] = v;
          if (out2) out2[(long long)e * o2E + (long long)r * o2Ld + col] = v;
        }
      }
    }
  }
}

// ---------------------------------------------------------------------------

extern "C" void kernel_launch(void* const* d_in, const int* in_sizes, int n_in,
                              void* d_out, int out_size, void* d_ws, size_t ws_size,
                              hipStream_t stream)
{
  (void)in_sizes; (void)n_in; (void)out_size; (void)ws_size;
  const float* x      = (const float*)d_in[0];
  const float* e0_W0  = (const float*)d_in[1];
  const float* e0_b0  = (const float*)d_in[2];
  const float* e0_W1  = (const float*)d_in[3];
  const float* e0_b1  = (const float*)d_in[4];
  const float* e0_W2  = (const float*)d_in[5];
  const float* e0_b2  = (const float*)d_in[6];
  const float* er_W0  = (const float*)d_in[7];
  const float* er_b0  = (const float*)d_in[8];
  const float* er_W1  = (const float*)d_in[9];
  const float* er_b1  = (const float*)d_in[10];
  const float* er_W2  = (const float*)d_in[11];
  const float* er_b2  = (const float*)d_in[12];
  const float* ff_W0  = (const float*)d_in[13];
  const float* ff_b0  = (const float*)d_in[14];
  const float* ff_W1  = (const float*)d_in[15];
  const float* ff_b1  = (const float*)d_in[16];
  const float* dec_W0 = (const float*)d_in[17];
  const float* dec_b0 = (const float*)d_in[18];
  const float* dec_Wh = (const float*)d_in[19];
  const float* dec_bh = (const float*)d_in[20];
  const float* dec_Wo = (const float*)d_in[21];
  const float* dec_bo = (const float*)d_in[22];
  float* outp = (float*)d_out;

  // ---- carve workspace ----------------------------------------------------
  char* ws = (char*)d_ws;
  size_t off = 0;
  auto aU = [&](size_t n) {
    void* p = ws + off; off = (off + n * 2 + 255) & ~(size_t)255;
    return (unsigned short*)p;
  };
  auto aF = [&](size_t n) {
    void* p = ws + off; off = (off + n * 4 + 255) & ~(size_t)255;
    return (float*)p;
  };
  unsigned short* sw_e0W0 = aU(16ull * 544 * 512);
  unsigned short* sw_e0W1 = aU(16ull * 512 * 512);
  unsigned short* sw_e0W2 = aU(16ull * 512 * 512);
  unsigned short* sw_erW0 = aU(48ull * 1024 * 512);
  unsigned short* sw_erW1 = aU(48ull * 512 * 512);
  unsigned short* sw_erW2 = aU(48ull * 512 * 512);
  unsigned short* sw_ffW0 = aU(16ull * 2048 * 2048);
  unsigned short* sw_ffW1 = aU(16ull * 2048 * 512);
  unsigned short* sw_dW0  = aU(9216ull * 2048);
  unsigned short* sw_dWh  = aU(4ull * 2048 * 2048);
  unsigned short* sw_dWo  = aU(2048ull * 1024);
  float* enc_cat = aF(16 * 4 * 2048);  // (E,B,NB*DM): block states, cat layout
  float* s_prev  = aF(16 * 4 * 512);   // (E,B,DM): previous block's output
  float* h0      = aF(16 * 4 * 512);
  float* h1      = aF(16 * 4 * 512);
  float* ffh     = aF(16 * 4 * 2048);
  float* dec_in  = aF(4 * 9216);       // (B, 16*512 + 1024); tail = dec state
  float* dh0     = aF(4 * 2048);
  float* dh1     = aF(4 * 2048);
  float* part    = aF(8 * 4 * 2048);   // split-K partials [S][4][N]

  // ---- weight conversion + swizzle (deterministic, redone every call) -----
  auto swz = [&](const float* W, unsigned short* dst, int K, int N, int nm) {
    long long total = (long long)(K / 32) * (N / 16) * 32;
    dim3 g((unsigned)((total + 255) / 256), (unsigned)nm);
    swz_kernel<<<g, 256, 0, stream>>>(W, dst, K, N, (long long)K * N);
  };
  swz(e0_W0, sw_e0W0,  544,  512, 16);
  swz(e0_W1, sw_e0W1,  512,  512, 16);
  swz(e0_W2, sw_e0W2,  512,  512, 16);
  swz(er_W0, sw_erW0, 1024,  512, 48);
  swz(er_W1, sw_erW1,  512,  512, 48);
  swz(er_W2, sw_erW2,  512,  512, 48);
  swz(ff_W0, sw_ffW0, 2048, 2048, 16);
  swz(ff_W1, sw_ffW1, 2048,  512, 16);
  swz(dec_W0, sw_dW0, 9216, 2048,  1);
  swz(dec_Wh, sw_dWh, 2048, 2048,  4);
  swz(dec_Wo, sw_dWo, 2048, 1024,  1);

  // ---- zero recurrent state ----------------------------------------------
  zero_f32_kernel<<<(16 * 4 * 2048 + 255) / 256, 256, 0, stream>>>(enc_cat, 16 * 4 * 2048);
  zero_f32_kernel<<<(4 * 9216 + 255) / 256, 256, 0, stream>>>(dec_in, 4 * 9216);

  auto gemm = [&](int nE,
                  const float* A0, long long a0e, int a0ld, int K0,
                  const float* A1, long long a1e, int a1ld, int K1,
                  const unsigned short* W, long long wE,
                  const float* bias, long long bE,
                  float* out,  long long oE,  int oLd,
                  float* out2, long long o2E, int o2Ld,
                  int N, int act, int splitK) {
    int z = (splitK > 0) ? (K0 + K1) / splitK : 1;
    dim3 g((unsigned)((N / 16 + 15) / 16), (unsigned)nE, (unsigned)z);
    gemm_bf16_kernel<<<g, 256, 0, stream>>>(A0, a0e, a0ld, K0, A1, a1e, a1ld, K1,
                                            W, wE, bias, bE, out, oE, oLd,
                                            out2, o2E, o2Ld, N, act, splitK);
  };
  auto reduceK = [&](int S, int N, const float* bias, float* out, int oLd,
                     float* out2, int o2Ld, int act) {
    reduce_kernel<<<(4 * N + 255) / 256, 256, 0, stream>>>(part, S, N, bias,
                                                           out, oLd, out2, o2Ld, act);
  };

  // ---- timestep scan ------------------------------------------------------
  for (int t = 0; t < 16; ++t) {
    // encoder block 0: cat[x_t(32), state0(512)] -> 512 (gelu,gelu,tanh)
    gemm(16, x + t * 32, 0, 512, 32, enc_cat, 8192, 2048, 512,
         sw_e0W0, 544 * 512, e0_b0, 512, h0, 2048, 512, nullptr, 0, 0, 512, 1, 0);
    gemm(16, h0, 2048, 512, 512, nullptr, 0, 0, 0,
         sw_e0W1, 512 * 512, e0_b1, 512, h1, 2048, 512, nullptr, 0, 0, 512, 1, 0);
    gemm(16, h1, 2048, 512, 512, nullptr, 0, 0, 0,
         sw_e0W2, 512 * 512, e0_b2, 512,
         s_prev, 2048, 512, enc_cat, 8192, 2048, 512, 2, 0);
    // encoder blocks 1..3: cat[s_prev(512), state_r(512)]
    for (int r = 1; r < 4; ++r) {
      int rr = r - 1;
      gemm(16, s_prev, 2048, 512, 512, enc_cat + r * 512, 8192, 2048, 512,
           sw_erW0 + (size_t)rr * 16 * 1024 * 512, 1024 * 512,
           er_b0 + rr * 16 * 512, 512, h0, 2048, 512, nullptr, 0, 0, 512, 1, 0);
      gemm(16, h0, 2048, 512, 512, nullptr, 0, 0, 0,
           sw_erW1 + (size_t)rr * 16 * 512 * 512, 512 * 512,
           er_b1 + rr * 16 * 512, 512, h1, 2048, 512, nullptr, 0, 0, 512, 1, 0);
      gemm(16, h1, 2048, 512, 512, nullptr, 0, 0, 0,
           sw_erW2 + (size_t)rr * 16 * 512 * 512, 512 * 512,
           er_b2 + rr * 16 * 512, 512,
           s_prev, 2048, 512, enc_cat + r * 512, 8192, 2048, 512, 2, 0);
    }
    // per-encoder FFNN over cat of 4 block states (2048 -> 2048 -> 512)
    gemm(16, enc_cat, 8192, 2048, 2048, nullptr, 0, 0, 0,
         sw_ffW0, 2048 * 2048, ff_b0, 2048, ffh, 8192, 2048, nullptr, 0, 0,
         2048, 1, 0);
    gemm(16, ffh, 8192, 2048, 2048, nullptr, 0, 0, 0,
         sw_ffW1, 2048 * 512, ff_b1, 512,
         dec_in, 512, 9216, nullptr, 0, 0, 512, 0, 0);  // write into (B, E*512)
    // decoder: 9216 -> 2048 (gelu) -> 4x 2048 (gelu) -> 1024 (tanh)
    // All decoder layers use split-K (S=8) for wave-level parallelism.
    gemm(1, dec_in, 0, 9216, 9216, nullptr, 0, 0, 0,
         sw_dW0, 0, nullptr, 0, part, 0, 2048, nullptr, 0, 0, 2048, 0, 9216 / 8);
    reduceK(8, 2048, dec_b0, dh0, 2048, nullptr, 0, 1);
    float* cur = dh0; float* nxt = dh1;
    for (int i = 0; i < 4; ++i) {
      gemm(1, cur, 0, 2048, 2048, nullptr, 0, 0, 0,
           sw_dWh + (size_t)i * 2048 * 2048, 0, nullptr, 0,
           part, 0, 2048, nullptr, 0, 0, 2048, 0, 2048 / 8);
      reduceK(8, 2048, dec_bh + i * 2048, nxt, 2048, nullptr, 0, 1);
      float* tmp = cur; cur = nxt; nxt = tmp;
    }
    gemm(1, cur, 0, 2048, 2048, nullptr, 0, 0, 0,
         sw_dWo, 0, nullptr, 0, part, 0, 1024, nullptr, 0, 0, 1024, 0, 2048 / 8);
    reduceK(8, 1024, dec_bo,
            outp + t * 1024, 16 * 1024,        // y[b, t, :]
            dec_in + 8192, 9216, 2);           // new decoder state
  }
}
